// LinearAttention_90623809945969
// MI455X (gfx1250) — matrix operand order repaired and checked by reading, not compiled, not run on hardware
//
#include <hip/hip_runtime.h>
#include <hip/hip_bf16.h>

// ---------------------------------------------------------------------------
// TransNormer-style linear attention block for MI455X (gfx1250, wave32).
// All matmuls via v_wmma_f32_16x16x32_bf16 (fp32->bf16 inputs, fp32 accum).
// Pure-copy tile staging uses GLOBAL_LOAD_ASYNC_TO_LDS_B128 (ASYNCcnt) when
// the toolchain exposes the builtin; falls back to uint4 copies otherwise.
// ---------------------------------------------------------------------------

typedef __attribute__((ext_vector_type(16))) __bf16 v16bf;
typedef __attribute__((ext_vector_type(8)))  float  v8f;

#define LDK1 40    // LDS row stride (elements) for 32-wide K tiles (+8 pad)
#define LDKQ 136   // LDS row stride (elements) for 128-wide D tiles (+8 pad)

// --- async global->LDS path (gfx1250) --------------------------------------
#if defined(__has_builtin)
#  if __has_builtin(__builtin_amdgcn_global_load_async_to_lds_b128)
#    define USE_ASYNC_LDS 1
#  endif
#  if __has_builtin(__builtin_amdgcn_s_wait_asynccnt)
#    define HAVE_WAIT_ASYNC_BUILTIN 1
#  endif
#endif

typedef int v4i_vs __attribute__((vector_size(16)));
typedef __attribute__((address_space(1))) v4i_vs* g_v4i_p;   // global src (non-const!)
typedef __attribute__((address_space(3))) v4i_vs* l_v4i_p;   // LDS dst

__device__ __forceinline__ void async_copy_b128(void* ldst, const void* gsrc) {
#ifdef USE_ASYNC_LDS
  __builtin_amdgcn_global_load_async_to_lds_b128(
      (g_v4i_p)gsrc, (l_v4i_p)ldst, 0, 0);
#else
  *(uint4*)ldst = *(const uint4*)gsrc;
#endif
}

__device__ __forceinline__ void async_wait() {
#ifdef USE_ASYNC_LDS
#  ifdef HAVE_WAIT_ASYNC_BUILTIN
  __builtin_amdgcn_s_wait_asynccnt(0);
#  else
  asm volatile("s_wait_asynccnt 0x0" ::: "memory");
#  endif
#endif
}
// ---------------------------------------------------------------------------

__device__ __forceinline__ unsigned short f2bf(float f) {
  union { float f; unsigned int u; } c; c.f = f;
  unsigned int u = c.u;
  u += 0x7FFFu + ((u >> 16) & 1u);   // round-to-nearest-even
  return (unsigned short)(u >> 16);
}
__device__ __forceinline__ unsigned int pk(unsigned short lo, unsigned short hi) {
  return (unsigned int)lo | ((unsigned int)hi << 16);
}

union FragU { v16bf v; uint4 q[2]; };

// A fragment, 16-bit 16x32 (MxK). lane: m = lane%16, half = lane/16.
// VGPR j<4 holds K = half*8 + 2j..+1 ; VGPR j>=4 holds K = 16 + half*8 + ...
// -> two contiguous 16B reads per lane from row-major [m][k] LDS.
__device__ __forceinline__ v16bf frag_a(const unsigned short* lds, int row, int ldk) {
  int lane = threadIdx.x & 31;
  int m = lane & 15, half = lane >> 4;
  const unsigned short* base = lds + (row + m) * ldk + half * 8;
  FragU u;
  u.q[0] = *(const uint4*)(base);
  u.q[1] = *(const uint4*)(base + 16);
  return u.v;
}

// B fragment, 16-bit 32x16 (KxN). lane: n = lane%16, half = lane/16.
// VGPR j<4 holds K = half*8 + 2j..+1 ; VGPR j>=4 holds K = 16 + half*8 + ... (same as A).
// LDS must be stored [n][k]-major (i.e. W^T / K-rows / V^T).
__device__ __forceinline__ v16bf frag_b(const unsigned short* lds, int ncol, int ldk) {
  int lane = threadIdx.x & 31;
  int n = lane & 15, half = lane >> 4;
  const unsigned short* base = lds + (ncol + n) * ldk + half * 8;
  FragU u;
  u.q[0] = *(const uint4*)(base);
  u.q[1] = *(const uint4*)(base + 16);
  return u.v;
}

__device__ __forceinline__ v8f wmma_bf(v16bf a, v16bf b, v8f c) {
  return __builtin_amdgcn_wmma_f32_16x16x32_bf16(
      false, a, false, b, (short)0, c, false, false);
}

// ---------------------------------------------------------------------------
// Kernel 1: C = X @ W  (M=4096, K=1024), tile 128x64, 8 waves (4x2), each wave
// computes a 32x32 block (2x2 WMMA tiles). Epilogue per mode:
//   0: silu + LRPE -> qrot[b,h,pos,0:64]=*cos, [64:128]=*sin   (bf16)
//   1: same -> krot
//   2: v transposed -> vT[b,h,dv,pos]                          (bf16)
//   3: g1[row,0:64] = X @ G1                                   (fp32)
// (fp32 sources convert to bf16 in flight -> stays on the VGPR staging path)
// ---------------------------------------------------------------------------
__global__ __launch_bounds__(256)
void proj_kernel(const float* __restrict__ X, const float* __restrict__ W,
                 const float* __restrict__ theta,
                 unsigned short* __restrict__ qk_dst,
                 unsigned short* __restrict__ vT,
                 float* __restrict__ g1,
                 int mode, int wstride)
{
  __shared__ __align__(16) unsigned short LX[128 * LDK1];
  __shared__ __align__(16) unsigned short LW[64 * LDK1];
  const int t = threadIdx.x;
  const int lane = t & 31, w = t >> 5;
  const int wm = w & 3, wn = w >> 2;
  const int rowbase = blockIdx.y * 128;
  const int colbase = blockIdx.x * 64;

  const v8f zf = {0.f, 0.f, 0.f, 0.f, 0.f, 0.f, 0.f, 0.f};
  v8f acc[2][2] = {{zf, zf}, {zf, zf}};

  for (int kt = 0; kt < 32; ++kt) {
    const int k0 = kt * 32;
    // stage X tile [128 m x 32 k] fp32 -> bf16, row-major
    {
      const int tk = t & 7, tm0 = t >> 3;
#pragma unroll
      for (int p = 0; p < 4; ++p) {
        const int m = tm0 + p * 32;
        const float4 xv = *(const float4*)(X + (size_t)(rowbase + m) * 1024 + k0 + tk * 4);
        *(uint2*)(LX + m * LDK1 + tk * 4) =
            make_uint2(pk(f2bf(xv.x), f2bf(xv.y)), pk(f2bf(xv.z), f2bf(xv.w)));
      }
    }
    // stage W^T tile [64 n x 32 k] fp32 -> bf16 (transposed on the fly)
    {
      const int tn = t & 15, kr0 = t >> 4;
#pragma unroll
      for (int p = 0; p < 2; ++p) {
        const int kr = kr0 + p * 16;
        const float4 wv = *(const float4*)(W + (size_t)(k0 + kr) * wstride + colbase + tn * 4);
        LW[(tn * 4 + 0) * LDK1 + kr] = f2bf(wv.x);
        LW[(tn * 4 + 1) * LDK1 + kr] = f2bf(wv.y);
        LW[(tn * 4 + 2) * LDK1 + kr] = f2bf(wv.z);
        LW[(tn * 4 + 3) * LDK1 + kr] = f2bf(wv.w);
      }
    }
    __syncthreads();
    v16bf a0 = frag_a(LX, wm * 32,      LDK1);
    v16bf a1 = frag_a(LX, wm * 32 + 16, LDK1);
    v16bf b0 = frag_b(LW, wn * 32,      LDK1);
    v16bf b1 = frag_b(LW, wn * 32 + 16, LDK1);
    acc[0][0] = wmma_bf(a0, b0, acc[0][0]);
    acc[0][1] = wmma_bf(a0, b1, acc[0][1]);
    acc[1][0] = wmma_bf(a1, b0, acc[1][0]);
    acc[1][1] = wmma_bf(a1, b1, acc[1][1]);
    __syncthreads();
  }

  // Epilogue. C/D layout: n = lane%16 (+ coltile), m = r + 8*(lane/16) (+ rowtile)
  const int half = lane >> 4, nl = lane & 15;
#pragma unroll
  for (int i = 0; i < 2; ++i)
#pragma unroll
    for (int j = 0; j < 2; ++j)
#pragma unroll
      for (int r = 0; r < 8; ++r) {
        const float val = acc[i][j][r];
        const int mg = rowbase + wm * 32 + i * 16 + r + half * 8;
        const int ng = colbase + wn * 32 + j * 16 + nl;
        const int b = mg >> 11, pos = mg & 2047;
        const int hh = ng >> 6, d = ng & 63;
        if (mode == 3) {
          g1[(size_t)mg * 64 + ng] = val;
        } else if (mode == 2) {
          vT[(size_t)((b * 16 + hh) * 64 + d) * 2048 + pos] = f2bf(val);
        } else {
          const float s  = val / (1.0f + __expf(-val));       // silu
          const float ang = theta[ng] * (float)pos;            // theta flat idx = h*64+d = ng
          const size_t base = (size_t)((b * 16 + hh) * 2048 + pos) * 128;
          qk_dst[base + d]      = f2bf(s * __cosf(ang));
          qk_dst[base + 64 + d] = f2bf(s * __sinf(ang));
        }
      }
}

// ---------------------------------------------------------------------------
// Kernel 2: causal-masked O = (Q K^T) V per (b,h). Query tile = 64 rows,
// key tiles of 32 streamed up the causal boundary. 8 waves: wm in 0..3 owns
// 16 query rows, wn in 0..1 owns 32 of the 64 value dims. All tile staging
// here is a pure bf16 copy -> async global->LDS data mover path.
// ---------------------------------------------------------------------------
__global__ __launch_bounds__(256)
void attn_kernel(const unsigned short* __restrict__ qrot,
                 const unsigned short* __restrict__ krot,
                 const unsigned short* __restrict__ vT,
                 unsigned short* __restrict__ aout)
{
  __shared__ __align__(16) unsigned short LQ [64 * LDKQ];   // [qrow][d0..127]
  __shared__ __align__(16) unsigned short LK [32 * LDKQ];   // [key ][d0..127]
  __shared__ __align__(16) unsigned short LVT[64 * LDK1];   // [dv  ][key0..31]
  __shared__ __align__(16) unsigned short LS [8 * 16 * LDK1]; // per-wave S scratch

  const int t = threadIdx.x;
  const int lane = t & 31, w = t >> 5;
  const int wm = w & 3, wn = w >> 2;
  const int half = lane >> 4, nl = lane & 15;
  const int bh = blockIdx.y;
  const int b = bh >> 4, h = bh & 15;
  const int qt = blockIdx.x;
  const int qrow0 = qt * 64;
  const size_t qkbase = (size_t)bh * 2048 * 128;

  // stage the full Q tile once: 64 x 128 bf16 (async copy)
#pragma unroll
  for (int p = 0; p < 4; ++p) {
    const int idx = p * 256 + t;
    const int m = idx >> 4, c8 = idx & 15;
    async_copy_b128(LQ + m * LDKQ + c8 * 8,
                    qrot + qkbase + (size_t)(qrow0 + m) * 128 + c8 * 8);
  }

  const v8f zf = {0.f, 0.f, 0.f, 0.f, 0.f, 0.f, 0.f, 0.f};
  v8f o0 = zf, o1 = zf;
  const int qm0 = wm * 16;
  unsigned short* myS = LS + w * 16 * LDK1;
  const int ntiles = 2 * qt + 2;   // key tiles covering keys <= qrow0+63
  async_wait();
  __syncthreads();

  for (int jt = 0; jt < ntiles; ++jt) {
    const int key0 = jt * 32;
    // stage K tile 32 x 128 bf16 (natural [key][d] == B-fragment major)
#pragma unroll
    for (int p = 0; p < 2; ++p) {
      const int idx = p * 256 + t;
      const int kr = idx >> 4, c8 = idx & 15;
      async_copy_b128(LK + kr * LDKQ + c8 * 8,
                      krot + qkbase + (size_t)(key0 + kr) * 128 + c8 * 8);
    }
    // stage V^T tile 64 dv x 32 keys bf16
    {
      const int dv = t >> 2, c = t & 3;
      async_copy_b128(LVT + dv * LDK1 + c * 8,
                      vT + (size_t)(bh * 64 + dv) * 2048 + key0 + c * 8);
    }
    async_wait();
    __syncthreads();

    // S = Q K^T : 16 q-rows x 32 keys, K-dim 128 = 4 WMMA steps
    v8f s0 = zf, s1 = zf;
#pragma unroll
    for (int d4 = 0; d4 < 4; ++d4) {
      v16bf aq  = frag_a(LQ + d4 * 32, qm0, LDKQ);
      v16bf bk0 = frag_b(LK + d4 * 32, 0,  LDKQ);
      v16bf bk1 = frag_b(LK + d4 * 32, 16, LDKQ);
      s0 = wmma_bf(aq, bk0, s0);
      s1 = wmma_bf(aq, bk1, s1);
    }

    // causal mask + fp32->bf16, re-layout C-frag -> A-frag via per-wave LDS
#pragma unroll
    for (int r = 0; r < 8; ++r) {
      const int ml = r + half * 8;
      const int qpos = qrow0 + qm0 + ml;
      const float v0 = (key0 + nl      <= qpos) ? s0[r] : 0.0f;
      const float v1 = (key0 + 16 + nl <= qpos) ? s1[r] : 0.0f;
      myS[ml * LDK1 + nl]      = f2bf(v0);
      myS[ml * LDK1 + 16 + nl] = f2bf(v1);
    }
    // O += S V (same-wave LDS RAW: hardware DScnt keeps it ordered)
    v16bf as  = frag_a(myS, 0, LDK1);
    v16bf bv0 = frag_b(LVT, wn * 32,      LDK1);
    v16bf bv1 = frag_b(LVT, wn * 32 + 16, LDK1);
    o0 = wmma_bf(as, bv0, o0);
    o1 = wmma_bf(as, bv1, o1);
    __syncthreads();
  }

  // write O as bf16 in [b, pos, h*64+dv] (input layout for the Wo GEMM)
#pragma unroll
  for (int r = 0; r < 8; ++r) {
    const int ml = r + half * 8;
    const size_t row = (size_t)(b * 2048 + qrow0 + qm0 + ml);
    aout[row * 1024 + h * 64 + wn * 32 + nl]      = f2bf(o0[r]);
    aout[row * 1024 + h * 64 + wn * 32 + 16 + nl] = f2bf(o1[r]);
  }
}

// ---------------------------------------------------------------------------
// Kernel 3: og = (A @ Wo) * sigmoid(g1 @ G2). Same 128x64 tiling; the gate
// GEMM (K=64) reuses the LDS buffers into a second accumulator. A tile is
// already bf16 -> async staging; weight tiles convert fp32->bf16 in VGPRs.
// ---------------------------------------------------------------------------
__global__ __launch_bounds__(256)
void outgate_kernel(const unsigned short* __restrict__ aout,
                    const float* __restrict__ Wo,
                    const float* __restrict__ g1,
                    const float* __restrict__ G2,
                    float* __restrict__ og)
{
  __shared__ __align__(16) unsigned short LA[128 * LDK1];
  __shared__ __align__(16) unsigned short LW[64 * LDK1];
  const int t = threadIdx.x;
  const int lane = t & 31, w = t >> 5;
  const int wm = w & 3, wn = w >> 2;
  const int rowbase = blockIdx.y * 128;
  const int colbase = blockIdx.x * 64;

  const v8f zf = {0.f, 0.f, 0.f, 0.f, 0.f, 0.f, 0.f, 0.f};
  v8f acc [2][2] = {{zf, zf}, {zf, zf}};
  v8f acc2[2][2] = {{zf, zf}, {zf, zf}};

  for (int kt = 0; kt < 32; ++kt) {
    const int k0 = kt * 32;
    // stage A tile (already bf16): 128 x 32 (async copy)
#pragma unroll
    for (int p = 0; p < 2; ++p) {
      const int idx = p * 256 + t;
      const int m = idx >> 2, c = idx & 3;
      async_copy_b128(LA + m * LDK1 + c * 8,
                      aout + (size_t)(rowbase + m) * 1024 + k0 + c * 8);
    }
    // stage Wo^T tile
    {
      const int tn = t & 15, kr0 = t >> 4;
#pragma unroll
      for (int p = 0; p < 2; ++p) {
        const int kr = kr0 + p * 16;
        const float4 wv = *(const float4*)(Wo + (size_t)(k0 + kr) * 1024 + colbase + tn * 4);
        LW[(tn * 4 + 0) * LDK1 + kr] = f2bf(wv.x);
        LW[(tn * 4 + 1) * LDK1 + kr] = f2bf(wv.y);
        LW[(tn * 4 + 2) * LDK1 + kr] = f2bf(wv.z);
        LW[(tn * 4 + 3) * LDK1 + kr] = f2bf(wv.w);
      }
    }
    async_wait();
    __syncthreads();
    v16bf a0 = frag_a(LA, wm * 32,      LDK1);
    v16bf a1 = frag_a(LA, wm * 32 + 16, LDK1);
    v16bf b0 = frag_b(LW, wn * 32,      LDK1);
    v16bf b1 = frag_b(LW, wn * 32 + 16, LDK1);
    acc[0][0] = wmma_bf(a0, b0, acc[0][0]);
    acc[0][1] = wmma_bf(a0, b1, acc[0][1]);
    acc[1][0] = wmma_bf(a1, b0, acc[1][0]);
    acc[1][1] = wmma_bf(a1, b1, acc[1][1]);
    __syncthreads();
  }

  // gate GEMM: g1[4096,64] @ G2[64,1024], K = 64 -> 2 steps
  for (int kt = 0; kt < 2; ++kt) {
    const int k0 = kt * 32;
    {
      const int tk = t & 7, tm0 = t >> 3;
#pragma unroll
      for (int p = 0; p < 4; ++p) {
        const int m = tm0 + p * 32;
        const float4 xv = *(const float4*)(g1 + (size_t)(rowbase + m) * 64 + k0 + tk * 4);
        *(uint2*)(LA + m * LDK1 + tk * 4) =
            make_uint2(pk(f2bf(xv.x), f2bf(xv.y)), pk(f2bf(xv.z), f2bf(xv.w)));
      }
    }
    {
      const int tn = t & 15, kr0 = t >> 4;
#pragma unroll
      for (int p = 0; p < 2; ++p) {
        const int kr = kr0 + p * 16;
        const float4 wv = *(const float4*)(G2 + (size_t)(k0 + kr) * 1024 + colbase + tn * 4);
        LW[(tn * 4 + 0) * LDK1 + kr] = f2bf(wv.x);
        LW[(tn * 4 + 1) * LDK1 + kr] = f2bf(wv.y);
        LW[(tn * 4 + 2) * LDK1 + kr] = f2bf(wv.z);
        LW[(tn * 4 + 3) * LDK1 + kr] = f2bf(wv.w);
      }
    }
    __syncthreads();
    v16bf a0 = frag_a(LA, wm * 32,      LDK1);
    v16bf a1 = frag_a(LA, wm * 32 + 16, LDK1);
    v16bf b0 = frag_b(LW, wn * 32,      LDK1);
    v16bf b1 = frag_b(LW, wn * 32 + 16, LDK1);
    acc2[0][0] = wmma_bf(a0, b0, acc2[0][0]);
    acc2[0][1] = wmma_bf(a0, b1, acc2[0][1]);
    acc2[1][0] = wmma_bf(a1, b0, acc2[1][0]);
    acc2[1][1] = wmma_bf(a1, b1, acc2[1][1]);
    __syncthreads();
  }

  const int half = lane >> 4, nl = lane & 15;
#pragma unroll
  for (int i = 0; i < 2; ++i)
#pragma unroll
    for (int j = 0; j < 2; ++j)
#pragma unroll
      for (int r = 0; r < 8; ++r) {
        const int mg = rowbase + wm * 32 + i * 16 + r + half * 8;
        const int ng = colbase + wn * 32 + j * 16 + nl;
        const float gate = 1.0f / (1.0f + __expf(-acc2[i][j][r]));
        og[(size_t)mg * 1024 + ng] = acc[i][j][r] * gate;
      }
}

// ---------------------------------------------------------------------------
// Kernel 4: per-row LayerNorm over 1024 features.
// ---------------------------------------------------------------------------
__global__ __launch_bounds__(256)
void ln_kernel(const float* __restrict__ og,
               const float* __restrict__ gamma,
               const float* __restrict__ beta,
               float* __restrict__ out)
{
  __shared__ float rs[256];
  __shared__ float rq[256];
  const int t = threadIdx.x;
  const size_t row = blockIdx.x;
  float v[4];
  float sum = 0.f, ss = 0.f;
#pragma unroll
  for (int k = 0; k < 4; ++k) {
    v[k] = og[row * 1024 + k * 256 + t];
    sum += v[k];
    ss  += v[k] * v[k];
  }
  rs[t] = sum; rq[t] = ss;
  __syncthreads();
  for (int off = 128; off > 0; off >>= 1) {
    if (t < off) { rs[t] += rs[t + off]; rq[t] += rq[t + off]; }
    __syncthreads();
  }
  const float mean = rs[0] * (1.0f / 1024.0f);
  const float var  = rq[0] * (1.0f / 1024.0f) - mean * mean;
  const float inv  = rsqrtf(var + 1e-5f);
#pragma unroll
  for (int k = 0; k < 4; ++k) {
    const int c = k * 256 + t;
    out[row * 1024 + c] = (v[k] - mean) * inv * gamma[c] + beta[c];
  }
}

// ---------------------------------------------------------------------------
extern "C" void kernel_launch(void* const* d_in, const int* in_sizes, int n_in,
                              void* d_out, int out_size, void* d_ws, size_t ws_size,
                              hipStream_t stream) {
  (void)in_sizes; (void)n_in; (void)out_size; (void)ws_size;
  const float* x     = (const float*)d_in[0];
  const float* Wq    = (const float*)d_in[1];
  const float* Wk    = (const float*)d_in[2];
  const float* Wv    = (const float*)d_in[3];
  const float* Wo    = (const float*)d_in[4];
  const float* G1    = (const float*)d_in[5];
  const float* G2    = (const float*)d_in[6];
  const float* gamma = (const float*)d_in[7];
  const float* beta  = (const float*)d_in[8];
  const float* theta = (const float*)d_in[9];

  char* ws = (char*)d_ws;
  unsigned short* qrot = (unsigned short*)ws;               ws += (size_t)2*16*2048*128*2; // 16 MB
  unsigned short* krot = (unsigned short*)ws;               ws += (size_t)2*16*2048*128*2; // 16 MB
  unsigned short* vT   = (unsigned short*)ws;               ws += (size_t)2*16*64*2048*2;  // 8 MB
  float*          g1f  = (float*)ws;                        ws += (size_t)4096*64*4;       // 1 MB
  unsigned short* aout = (unsigned short*)ws;               ws += (size_t)4096*1024*2;     // 8 MB
  float*          og   = (float*)ws;                        /* 16 MB */
  float*          outp = (float*)d_out;

  const dim3 blk(256);
  // q / k / v / gate-pre projections
  proj_kernel<<<dim3(16, 32), blk, 0, stream>>>(x, Wq, theta, qrot, vT, g1f, 0, 1024);
  proj_kernel<<<dim3(16, 32), blk, 0, stream>>>(x, Wk, theta, krot, vT, g1f, 1, 1024);
  proj_kernel<<<dim3(16, 32), blk, 0, stream>>>(x, Wv, theta, qrot, vT, g1f, 2, 1024);
  proj_kernel<<<dim3(1,  32), blk, 0, stream>>>(x, G1, theta, qrot, vT, g1f, 3, 64);
  // causal linear attention per (b,h)
  attn_kernel<<<dim3(32, 32), blk, 0, stream>>>(qrot, krot, vT, aout);
  // output projection + gate
  outgate_kernel<<<dim3(16, 32), blk, 0, stream>>>(aout, Wo, g1f, G2, og);
  // layernorm
  ln_kernel<<<dim3(4096), blk, 0, stream>>>(og, gamma, beta, outp);
}
